// NgramDiversityLoss_36455682408762
// MI455X (gfx1250) — matrix operand order repaired and checked by reading, not compiled, not run on hardware
//
#include <hip/hip_runtime.h>
#include <hip/hip_bf16.h>

typedef __attribute__((ext_vector_type(16))) _Float16 v16h;
typedef __attribute__((ext_vector_type(8)))  _Float16 v8h;
typedef __attribute__((ext_vector_type(8)))  float    v8f;

#define NVOC 512
#define NBAT 8
#define NLEN 2048
#define NROW (NBAT * NLEN)      /* 16384 rows of p */
#define WPAD NROW               /* pT row width (multiple of 8 -> 16B aligned b128) */
#define EPSF 1e-10f
#define LOGEPSF (-23.025850929940457f)

// WGP-scope prefetch: no scope: modifier => SCOPE 0 (WGP) => fills all cache levels.
__device__ __forceinline__ void ndl_prefetch_wgp(const void* p) {
  unsigned long long a = (unsigned long long)p;
  asm volatile("global_prefetch_b8 %0, off" :: "v"(a));
}

// ---------------- deterministic block reductions (256 threads) ----------------
__device__ __forceinline__ float blk_sum256(float v, float* red) {
  int t = threadIdx.x;
  red[t] = v; __syncthreads();
  #pragma unroll
  for (int s = 128; s > 0; s >>= 1) {
    if (t < s) red[t] += red[t + s];
    __syncthreads();
  }
  float r = red[0]; __syncthreads();
  return r;
}
__device__ __forceinline__ float blk_max256(float v, float* red) {
  int t = threadIdx.x;
  red[t] = v; __syncthreads();
  #pragma unroll
  for (int s = 128; s > 0; s >>= 1) {
    if (t < s) red[t] = fmaxf(red[t], red[t + s]);
    __syncthreads();
  }
  float r = red[0]; __syncthreads();
  return r;
}

__device__ __forceinline__ float fixv(float v) {
  if (__builtin_isnan(v)) return 0.f;
  if (__builtin_isinf(v)) return v > 0.f ? 80.f : -80.f;
  return v;
}

// ---------------- 1) softmax -> transposed fp16 p (pT) and shifted copy (pT2) -
// pT [c, r]  = p[r, c]          (512 x 16384)
// pT2[c, q]  = p[q+1, c]        (512 x 16384), col 16383 = zero pad
__global__ __launch_bounds__(256) void ndl_softmax_T_kernel(
    const float* __restrict__ logits, _Float16* __restrict__ pT,
    _Float16* __restrict__ pT2) {
  __shared__ float red[256];
  int r = blockIdx.x, t = threadIdx.x;
  if (r >= NROW) {                 // single pad block: virtual row NROW -> zeros
    for (int c = t; c < NVOC; c += 256)
      pT2[(size_t)c * WPAD + (NROW - 1)] = (_Float16)0.0f;
    return;
  }
  const float* x = logits + (size_t)r * NVOC;
  float v0 = fixv(x[t]);
  float v1 = fixv(x[t + 256]);
  float mx = blk_max256(fmaxf(v0, v1), red);
  float e0 = expf(v0 - mx), e1 = expf(v1 - mx);
  float s = blk_sum256(e0 + e1, red);
  float inv = 1.f / s;
  _Float16 h0 = (_Float16)(e0 * inv);
  _Float16 h1 = (_Float16)(e1 * inv);
  pT[(size_t)t * WPAD + r] = h0;
  pT[(size_t)(t + 256) * WPAD + r] = h1;
  if (r >= 1) {
    pT2[(size_t)t * WPAD + (r - 1)] = h0;
    pT2[(size_t)(t + 256) * WPAD + (r - 1)] = h1;
  }
}

// ---------------- 2) third-position marginal m[c] = sum_{n, l>=2} p[n,l,c] ----
__global__ __launch_bounds__(256) void ndl_marginal_kernel(
    const _Float16* __restrict__ pT, float* __restrict__ m) {
  __shared__ float red[256];
  int c = blockIdx.x, t = threadIdx.x;
  const _Float16* row = pT + (size_t)c * WPAD;
  float s = 0.f;
  for (int r = t; r < NROW; r += 256)
    if ((r & (NLEN - 1)) >= 2) s += (float)row[r];
  float tot = blk_sum256(s, red);
  if (t == 0) m[c] = tot;
}

// ---------------- 3) WMMA GEMM: G[a,b] = sum_r p[r,a] * p[r+1,b] --------------
// A 16x32 f16 fragment: lane(m=lane&15): K runs {0-7,16-23} (lanes<16) / {8-15,24-31}
__device__ __forceinline__ v16h ndl_load_afrag(const _Float16* __restrict__ pT,
                                               int a0, int k0, int lane) {
  int h = lane >> 4;
  const _Float16* p = pT + (size_t)(a0 + (lane & 15)) * WPAD + k0 + h * 8;
  v8h lo = *(const v8h*)p;          // K = h?8..15 : 0..7
  v8h hi = *(const v8h*)(p + 16);   // K = h?24..31: 16..23
  return __builtin_shufflevector(lo, hi, 0, 1, 2, 3, 4, 5, 6, 7,
                                 8, 9, 10, 11, 12, 13, 14, 15);
}
// B 32x16 f16 fragment: lane(n=lane&15): K runs {0-7,16-23} (lanes<16) / {8-15,24-31}
__device__ __forceinline__ v16h ndl_load_bfrag(const _Float16* __restrict__ pT2,
                                               int b0, int k0, int lane) {
  int h = lane >> 4;
  const _Float16* p = pT2 + (size_t)(b0 + (lane & 15)) * WPAD + k0 + h * 8;
  v8h lo = *(const v8h*)p;          // K = h?8..15 : 0..7
  v8h hi = *(const v8h*)(p + 16);   // K = h?24..31: 16..23
  return __builtin_shufflevector(lo, hi, 0, 1, 2, 3, 4, 5, 6, 7,
                                 8, 9, 10, 11, 12, 13, 14, 15);
}

#define WMMA_F16(A, B, C) \
  __builtin_amdgcn_wmma_f32_16x16x32_f16(false, (A), false, (B), (short)0, (C), false, false)

// grid (4, 8, 8): x -> 128-wide col tiles, y -> 64-tall row tiles, z -> K split.
// Each 256-thread block = 8 waves in 2x4; each wave owns a 32x32 output tile.
__global__ __launch_bounds__(256) void ndl_gemm_wmma_kernel(
    const _Float16* __restrict__ pT, const _Float16* __restrict__ pT2,
    float* __restrict__ Gp) {
  int lane = threadIdx.x & 31;
  int wave = threadIdx.x >> 5;
  int a0 = blockIdx.y * 64 + (wave >> 2) * 32;
  int b0 = blockIdx.x * 128 + (wave & 3) * 32;
  int kbeg = blockIdx.z * (NROW / 8);
  int kend = kbeg + (NROW / 8);
  v8f acc00 = {}, acc01 = {}, acc10 = {}, acc11 = {};
  for (int k0 = kbeg; k0 < kend; k0 += 32) {
    v16h A0 = ndl_load_afrag(pT, a0, k0, lane);
    v16h A1 = ndl_load_afrag(pT, a0 + 16, k0, lane);
    v16h B0 = ndl_load_bfrag(pT2, b0, k0, lane);
    v16h B1 = ndl_load_bfrag(pT2, b0 + 16, k0, lane);
    if (k0 + 32 < kend) {  // WGP-scope prefetch of next K slab into near caches
      ndl_prefetch_wgp(pT + (size_t)(a0 + (lane & 15)) * WPAD + k0 + 32);
      ndl_prefetch_wgp(pT2 + (size_t)(b0 + (lane & 15)) * WPAD + k0 + 32);
    }
    acc00 = WMMA_F16(A0, B0, acc00);
    acc01 = WMMA_F16(A0, B1, acc01);
    acc10 = WMMA_F16(A1, B0, acc10);
    acc11 = WMMA_F16(A1, B1, acc11);
  }
  float* Gz = Gp + (size_t)blockIdx.z * NVOC * NVOC;  // private partial: deterministic
  int col = lane & 15;
  int rofs = (lane >> 4) * 8;
  #pragma unroll
  for (int i = 0; i < 8; i++) {
    int ra = a0 + rofs + i;
    int rb = ra + 16;
    Gz[(size_t)ra * NVOC + b0 + col]      = acc00[i];
    Gz[(size_t)ra * NVOC + b0 + 16 + col] = acc01[i];
    Gz[(size_t)rb * NVOC + b0 + col]      = acc10[i];
    Gz[(size_t)rb * NVOC + b0 + 16 + col] = acc11[i];
  }
}

// ---------------- 4) fold K-partials + rank-7 / rank-8 corrections ------------
__global__ __launch_bounds__(256) void ndl_correct_kernel(
    const _Float16* __restrict__ pT, const float* __restrict__ Gp,
    float* __restrict__ bigram, float* __restrict__ bipart) {
  int idx = blockIdx.x * 256 + threadIdx.x;
  int a = idx >> 9, b = idx & (NVOC - 1);
  float g = 0.f;
  #pragma unroll
  for (int z = 0; z < 8; z++) g += Gp[(size_t)z * NVOC * NVOC + idx];
  float cross = 0.f;  // remove cross-batch pairs (rows n*L+2047 , n*L+2048)
  #pragma unroll
  for (int n = 0; n < NBAT - 1; n++) {
    int r1 = n * NLEN + NLEN - 1;
    cross += (float)pT[(size_t)a * WPAD + r1] * (float)pT[(size_t)b * WPAD + r1 + 1];
  }
  float bg = g - cross;
  float lastp = 0.f;  // bi_part drops each batch's (L-2, L-1) pair
  #pragma unroll
  for (int n = 0; n < NBAT; n++) {
    int r1 = n * NLEN + NLEN - 2;
    lastp += (float)pT[(size_t)a * WPAD + r1] * (float)pT[(size_t)b * WPAD + r1 + 1];
  }
  bigram[idx] = bg;
  bipart[idx] = bg - lastp;
}

// ---------------- 5) partial sums of bigram / bipart --------------------------
__global__ __launch_bounds__(256) void ndl_sums_kernel(
    const float* __restrict__ bigram, const float* __restrict__ bipart,
    float* __restrict__ pSb, float* __restrict__ pSp) {
  __shared__ float red[256];
  int t = threadIdx.x;
  size_t base = (size_t)blockIdx.x * 1024;
  float sb = 0.f, sp = 0.f;
  for (int i = t; i < 1024; i += 256) { sb += bigram[base + i]; sp += bipart[base + i]; }
  sb = blk_sum256(sb, red);
  sp = blk_sum256(sp, red);
  if (t == 0) { pSb[blockIdx.x] = sb; pSp[blockIdx.x] = sp; }
}

__global__ __launch_bounds__(256) void ndl_reduce_sums_kernel(
    const float* __restrict__ pSb, const float* __restrict__ pSp,
    const float* __restrict__ m, float* __restrict__ scal) {
  __shared__ float red[256];
  int t = threadIdx.x;
  float v = blk_sum256(pSb[t], red);
  if (t == 0) scal[0] = v;                 // S_bigram
  float w = blk_sum256(pSp[t], red);
  if (t == 0) scal[1] = w;                 // S_bipart
  float u = blk_sum256(m[t] + m[t + 256], red);
  if (t == 0) scal[2] = u;                 // S_marginal
}

// ---------------- 6) entropies: bigram directly, trigram via factorization ----
__global__ __launch_bounds__(256) void ndl_entropy_kernel(
    const float* __restrict__ bigram, const float* __restrict__ bipart,
    const float* __restrict__ m, const float* __restrict__ scal,
    float* __restrict__ pHb, float* __restrict__ pHt) {
  __shared__ float sm[NVOC];
  __shared__ float slm[NVOC];
  __shared__ float red[256];
  int t = threadIdx.x;
  for (int c = t; c < NVOC; c += 256) {
    float mv = m[c];
    sm[c] = mv;
    slm[c] = (mv > 0.f) ? logf(mv) : 0.f;
  }
  __syncthreads();
  float invDbi = 1.f / (scal[0] + EPSF);
  float invDtri = 1.f / (scal[1] * scal[2] + EPSF);
  int idx = blockIdx.x * 256 + t;          // one (a,b) per thread
  float hbi = 0.f, htri = 0.f;
  float tbi = bigram[idx] * invDbi;
  if (tbi > 0.f) hbi = -tbi * logf(fmaxf(tbi, EPSF));
  float u = bipart[idx] * invDtri;         // tri[a,b,c] = u * m[c]
  if (u > 0.f) {
    float lu = logf(u);
    for (int c = 0; c < NVOC; c++) {       // per-element EPS clip, no logs inside
      float tv = u * sm[c];
      float lt = (tv >= EPSF) ? (lu + slm[c]) : LOGEPSF;
      htri -= tv * lt;                     // tv==0 -> xlogy contributes 0
    }
  }
  hbi = blk_sum256(hbi, red);
  htri = blk_sum256(htri, red);
  if (t == 0) { pHb[blockIdx.x] = hbi; pHt[blockIdx.x] = htri; }
}

// ---------------- 7) final combine --------------------------------------------
__global__ __launch_bounds__(256) void ndl_final_kernel(
    const float* __restrict__ pHb, const float* __restrict__ pHt,
    float* __restrict__ out) {
  __shared__ float red[256];
  int t = threadIdx.x;
  float hb = 0.f, ht = 0.f;
  #pragma unroll
  for (int i = 0; i < 4; i++) { hb += pHb[t + 256 * i]; ht += pHt[t + 256 * i]; }
  hb = blk_sum256(hb, red);
  ht = blk_sum256(ht, red);
  if (t == 0) {
    float max_log_bi = 2.f * logf(512.f);
    float max_log_tri = 3.f * logf(512.f);
    float sb = 1.f - hb / (max_log_bi + EPSF);
    sb = fminf(fmaxf(sb, 0.f), 1.f);
    float st = 1.f - ht / (max_log_tri + EPSF);
    st = fminf(fmaxf(st, 0.f), 1.f);
    float tot = 0.5f * sb + 0.5f * st;
    if (__builtin_isnan(tot)) tot = 0.f;
    if (__builtin_isinf(tot)) tot = tot > 0.f ? 1.f : 0.f;
    out[0] = fminf(fmaxf(tot, 0.f), 1.f);
  }
}

extern "C" void kernel_launch(void* const* d_in, const int* in_sizes, int n_in,
                              void* d_out, int out_size, void* d_ws, size_t ws_size,
                              hipStream_t stream) {
  (void)in_sizes; (void)n_in; (void)out_size; (void)ws_size;
  const float* logits = (const float*)d_in[0];
  float* out = (float*)d_out;

  // workspace layout (~42 MiB)
  _Float16* pT  = (_Float16*)d_ws;                         // 512 * 16384 f16
  _Float16* pT2 = pT + (size_t)NVOC * WPAD;                // 512 * 16384 f16
  float* Gp     = (float*)(pT2 + (size_t)NVOC * WPAD);     // 8 * 512*512 f32
  float* bigram = Gp + (size_t)8 * NVOC * NVOC;            // 512*512
  float* bipart = bigram + (size_t)NVOC * NVOC;            // 512*512
  float* m      = bipart + (size_t)NVOC * NVOC;            // 512
  float* pSb    = m + NVOC;                                // 256
  float* pSp    = pSb + 256;                               // 256
  float* pHb    = pSp + 256;                               // 1024
  float* pHt    = pHb + 1024;                              // 1024
  float* scal   = pHt + 1024;                              // 3

  hipLaunchKernelGGL(ndl_softmax_T_kernel, dim3(NROW + 1), dim3(256), 0, stream,
                     logits, pT, pT2);
  hipLaunchKernelGGL(ndl_marginal_kernel, dim3(NVOC), dim3(256), 0, stream, pT, m);
  hipLaunchKernelGGL(ndl_gemm_wmma_kernel, dim3(4, 8, 8), dim3(256), 0, stream,
                     pT, pT2, Gp);
  hipLaunchKernelGGL(ndl_correct_kernel, dim3(NVOC * NVOC / 256), dim3(256), 0,
                     stream, pT, Gp, bigram, bipart);
  hipLaunchKernelGGL(ndl_sums_kernel, dim3(256), dim3(256), 0, stream,
                     bigram, bipart, pSb, pSp);
  hipLaunchKernelGGL(ndl_reduce_sums_kernel, dim3(1), dim3(256), 0, stream,
                     pSb, pSp, m, scal);
  hipLaunchKernelGGL(ndl_entropy_kernel, dim3(NVOC * NVOC / 256), dim3(256), 0,
                     stream, bigram, bipart, m, scal, pHb, pHt);
  hipLaunchKernelGGL(ndl_final_kernel, dim3(1), dim3(256), 0, stream,
                     pHb, pHt, out);
}
